// BeamSearchInference_10428180595041
// MI455X (gfx1250) — matrix operand order repaired and checked by reading, not compiled, not run on hardware
//
#include <hip/hip_runtime.h>

// ---------------------------------------------------------------------------
// CDNA5 / gfx1250 beam-search decoder.
// wave32, WMMA bf16 16x16x32 with fp32 accumulate for every GEMM.
// ---------------------------------------------------------------------------

typedef __bf16 bf16_t;
typedef __attribute__((ext_vector_type(16))) __bf16 v16bf;
typedef __attribute__((ext_vector_type(8)))  float  v8f;

#define BATCH   32
#define SLEN    256
#define ENC     512
#define ATT     128
#define NS      512
#define EMB     128
#define NCLS    173
#define NCLSP   176
#define BEAM    4
#define MAXLEN  50
#define NHAN    11172
#define NCHARS  11272
#define KXH     1152     // [glimpse(512) | emb(128) | state(512)]
#define NG      1536

__device__ __forceinline__ float sigm(float x) { return 1.0f / (1.0f + __expf(-x)); }
__device__ __forceinline__ float logclip(float x) {
  x = fminf(fmaxf(x, 1e-20f), 1.0f - 1e-20f);
  return __logf(x);
}

__device__ __forceinline__ v8f wmma_bf(v16bf a, v16bf b, v8f c) {
  return __builtin_amdgcn_wmma_f32_16x16x32_bf16(false, a, false, b, (short)0, c, false, false);
}

// A tile (16x32 bf16) from row-major X[rows][ld], rows >= "rows" are zero.
// ISA layout: lanes 0-15 row M=lane, halves h<8 -> K=kbase+h, h>=8 -> K=kbase+16+(h-8);
// lanes 16-31 same row, K offset +8.
__device__ __forceinline__ v16bf ldA(const bf16_t* X, int ld, int rows, int kbase, int lane) {
  int m  = lane & 15;
  int kh = (lane >> 4) << 3;
  v16bf a;
#pragma unroll
  for (int h = 0; h < 16; ++h) {
    int k = kbase + ((h >> 3) << 4) + kh + (h & 7);
    a[h] = (m < rows) ? X[(size_t)m * ld + k] : (bf16_t)0.0f;
  }
  return a;
}

// B tile (32x16 bf16) from TRANSPOSED weights WT[N][ldk]: lane n = lane&15,
// same K order as ldA: h<8 -> K=kbase+kh+h, h>=8 -> K=kbase+16+kh+(h-8), kh = lane>=16 ? 8 : 0.
__device__ __forceinline__ v16bf ldBT(const bf16_t* WT, int ldk, int nbase, int kbase, int lane) {
  const bf16_t* p = WT + (size_t)(nbase + (lane & 15)) * ldk + kbase + ((lane >> 4) << 3);
  v16bf b;
#pragma unroll
  for (int h = 0; h < 16; ++h) b[h] = p[((h >> 3) << 4) + (h & 7)];
  return b;
}

// ------------------------------- prep kernels -------------------------------

__global__ void k_cvt_enc(const float* __restrict__ enc,
                          bf16_t* __restrict__ enc_bf, bf16_t* __restrict__ encT) {
  const int total = BATCH * SLEN * ENC;
  for (int i = blockIdx.x * blockDim.x + threadIdx.x; i < total; i += blockDim.x * gridDim.x) {
    int c = i & 511;
    int s = (i >> 9) & 255;
    int bb = i >> 17;
    float v = enc[i];
    enc_bf[i] = (bf16_t)v;
    encT[((size_t)bb * ENC + c) * SLEN + s] = (bf16_t)v;  // [b][c][s]
  }
}

__global__ void k_transpose(const float* __restrict__ src, bf16_t* __restrict__ dst, int K, int N) {
  int total = K * N;
  for (int i = blockIdx.x * blockDim.x + threadIdx.x; i < total; i += blockDim.x * gridDim.x) {
    int n = i / K, k = i - n * K;
    dst[i] = (bf16_t)src[k * N + n];
  }
}

// stacked [Wx;Wh] transposed: dst[n][k], k<640 -> Wx[k][n], else Wh[k-640][n]
__global__ void k_wxh(const float* __restrict__ Wx, const float* __restrict__ Wh,
                      bf16_t* __restrict__ dst) {
  int total = NG * KXH;
  for (int i = blockIdx.x * blockDim.x + threadIdx.x; i < total; i += blockDim.x * gridDim.x) {
    int n = i / KXH, k = i - n * KXH;
    float v = (k < 640) ? Wx[k * NG + n] : Wh[(k - 640) * NG + n];
    dst[i] = (bf16_t)v;
  }
}

__global__ void k_wc(const float* __restrict__ Wc, bf16_t* __restrict__ dst) {
  int total = NCLSP * NS;
  for (int i = blockIdx.x * blockDim.x + threadIdx.x; i < total; i += blockDim.x * gridDim.x) {
    int n = i / NS, k = i - n * NS;
    dst[i] = (n < NCLS) ? (bf16_t)Wc[k * NCLS + n] : (bf16_t)0.0f;
  }
}

// key[b,s,:] = enc[b,s,:] @ Wk + bk   (M=8192 rows, K=512, N=128) — all WMMA.
__global__ void __launch_bounds__(256) k_keys(const bf16_t* __restrict__ enc_bf,
                                              const bf16_t* __restrict__ WkT,
                                              const float* __restrict__ bk,
                                              float* __restrict__ key) {
  int rowbase = blockIdx.x * 16;
  int wave = threadIdx.x >> 5, lane = threadIdx.x & 31;
  int nbase = wave * 16;
  v8f acc = {};
  for (int kb = 0; kb < ENC; kb += 32) {
    v16bf a = ldA(enc_bf + (size_t)rowbase * ENC, ENC, 16, kb, lane);
    v16bf bm = ldBT(WkT, ENC, nbase, kb, lane);
    acc = wmma_bf(a, bm, acc);
  }
  int n = nbase + (lane & 15);
  int mb = (lane >> 4) << 3;
#pragma unroll
  for (int r = 0; r < 8; ++r) {
    int m = mb + r;
    key[(size_t)(rowbase + m) * ATT + n] = acc[r] + bk[n];
  }
}

// ------------------------------ decode kernel -------------------------------
// One block per batch element; whole 50-step scan runs inside the block.

__global__ void __launch_bounds__(256) k_decode(
    const float* __restrict__ key, const bf16_t* __restrict__ encT,
    const float* __restrict__ embed, const int* __restrict__ maskg,
    const bf16_t* __restrict__ WqT, const float* __restrict__ bq,
    const float* __restrict__ Wsv, const float* __restrict__ bsv,
    const bf16_t* __restrict__ WxhT, const float* __restrict__ bg,
    const bf16_t* __restrict__ WcT, const float* __restrict__ bc,
    int* __restrict__ ids_tr, int* __restrict__ uni_tr,
    float* __restrict__ flogp, int* __restrict__ flen) {
  const int b = blockIdx.x;
  const int tid = threadIdx.x;
  const int wave = tid >> 5, lane = tid & 31;

  __shared__ float  s_state[BEAM][NS];
  __shared__ float  s_nstate[BEAM][NS];
  __shared__ bf16_t s_xin[BEAM][KXH];     // [glimpse | emb | state] bf16
  __shared__ float  s_probs[BEAM][NCLSP];
  __shared__ float  s_red2[256];
  __shared__ float  s_red[16];
  __shared__ int    s_uni[BEAM];
  __shared__ float  s_logp[BEAM];
  __shared__ int    s_len[BEAM];
  __shared__ int    s_fin[BEAM];
  __shared__ int    s_pb[BEAM];
  // Phase-overlaid union region (32 KB):
  //  phase score  : s_score @0 (4K), s_attb @16K (2K), s_q @20K (2K)
  //  phase gates  : s_gzr @0 (16K), s_xn @16K (8K), s_hn @24K (8K)
  //  phase probs  : s_nsbf @0 (4K)
  //  phase top-k  : s_cval @8K (4K), s_cidx @12K (4K)
  __shared__ __align__(16) char u_buf[32768];
  float  (*s_score)[SLEN] = (float  (*)[SLEN])(u_buf);
  bf16_t (*s_attb)[SLEN]  = (bf16_t (*)[SLEN])(u_buf + 16384);
  float  (*s_q)[ATT]      = (float  (*)[ATT]) (u_buf + 20480);
  float  (*s_gzr)[1024]   = (float  (*)[1024])(u_buf);
  float  (*s_xn)[NS]      = (float  (*)[NS])  (u_buf + 16384);
  float  (*s_hn)[NS]      = (float  (*)[NS])  (u_buf + 24576);
  bf16_t (*s_nsbf)[NS]    = (bf16_t (*)[NS])  (u_buf);
  float* s_cval           = (float*)(u_buf + 8192);
  int*   s_cidx           = (int*)  (u_buf + 12288);

  for (int i = tid; i < BEAM * NS; i += 256) s_state[i >> 9][i & 511] = 0.0f;
  if (tid < BEAM) { s_uni[tid] = 10; s_logp[tid] = 0.0f; s_len[tid] = 0; s_fin[tid] = 0; }
  __syncthreads();

  const float*  keyb  = key  + (size_t)b * SLEN * ATT;
  const bf16_t* encTb = encT + (size_t)b * ENC * SLEN;

  for (int t = 0; t < MAXLEN; ++t) {
    // (a) state -> bf16 tail of xin
    for (int i = tid; i < BEAM * NS; i += 256) {
      int j = i >> 9, c = i & 511;
      s_xin[j][640 + c] = (bf16_t)s_state[j][c];
    }
    __syncthreads();

    // (b) q = state @ Wq + bq   (wave w owns one 16-col tile)
    {
      v8f acc = {};
      for (int kb = 0; kb < NS; kb += 32) {
        v16bf a  = ldA(&s_xin[0][0], KXH, BEAM, 640 + kb, lane);
        v16bf bm = ldBT(WqT, NS, wave * 16, kb, lane);
        acc = wmma_bf(a, bm, acc);
      }
      int n = wave * 16 + (lane & 15);
      int mb = (lane >> 4) << 3;
#pragma unroll
      for (int r = 0; r < 8; ++r) {
        int m = mb + r;
        if (m < BEAM) s_q[m][n] = acc[r] + bq[n];
      }
    }
    __syncthreads();

    // (c) score[j][s] = tanh(key+q) . Ws + bs - mask_add
    for (int i = tid; i < BEAM * SLEN; i += 256) {
      int j = i >> 8, sv = i & 255;
      const float* kp = keyb + (size_t)sv * ATT;
      float acc = bsv[0];
      for (int c = 0; c < ATT; ++c) acc += tanhf(kp[c] + s_q[j][c]) * Wsv[c];
      float madd = (1.0f - (float)maskg[b * SLEN + sv]) * 4.6566128731e-10f;  // 2^-31
      s_score[j][sv] = acc - madd;
    }
    __syncthreads();

    // (d) softmax over S per beam -> bf16 attention
    {
      int j = tid >> 6, g = tid & 63;
      float lm = -3.4e38f;
      for (int sv = g; sv < SLEN; sv += 64) lm = fmaxf(lm, s_score[j][sv]);
      s_red2[tid] = lm;
      __syncthreads();
      if (g == 0) {
        float m = s_red2[tid];
        for (int x = 1; x < 64; ++x) m = fmaxf(m, s_red2[tid + x]);
        s_red[j] = m;
      }
      __syncthreads();
      float mx = s_red[j], ls = 0.0f;
      for (int sv = g; sv < SLEN; sv += 64) ls += __expf(s_score[j][sv] - mx);
      s_red2[tid] = ls;
      __syncthreads();
      if (g == 0) {
        float sm = 0.0f;
        for (int x = 0; x < 64; ++x) sm += s_red2[tid + x];
        s_red[8 + j] = sm;
      }
      __syncthreads();
      float inv = 1.0f / s_red[8 + j];
      for (int sv = g; sv < SLEN; sv += 64)
        s_attb[j][sv] = (bf16_t)(__expf(s_score[j][sv] - mx) * inv);
    }
    __syncthreads();

    // (e) glimpse = att @ enc  -> xin[:,0:512]  (wave w owns 4 tiles)
    for (int it = 0; it < 4; ++it) {
      int nbase = (wave + 8 * it) * 16;
      v8f acc = {};
      for (int kb = 0; kb < SLEN; kb += 32) {
        v16bf a  = ldA(&s_attb[0][0], SLEN, BEAM, kb, lane);
        v16bf bm = ldBT(encTb, SLEN, nbase, kb, lane);
        acc = wmma_bf(a, bm, acc);
      }
      int n = nbase + (lane & 15);
      int mb = (lane >> 4) << 3;
#pragma unroll
      for (int r = 0; r < 8; ++r) {
        int m = mb + r;
        if (m < BEAM) s_xin[m][n] = (bf16_t)acc[r];
      }
    }
    // (f) embedding gather -> xin[:,512:640]
    for (int i = tid; i < BEAM * EMB; i += 256) {
      int j = i >> 7, e = i & 127;
      int u = s_uni[j];
      u = u < 0 ? 0 : (u > 65535 ? 65535 : u);
      s_xin[j][512 + e] = (bf16_t)embed[(size_t)u * EMB + e];
    }
    __syncthreads();

    // (g) fused z|r gates: [ctx|h] @ [Wx;Wh][:, 0:1024]
    for (int nt = 0; nt < 8; ++nt) {
      int nbase = (wave * 8 + nt) * 16;
      v8f acc = {};
      for (int kb = 0; kb < KXH; kb += 32) {
        v16bf a  = ldA(&s_xin[0][0], KXH, BEAM, kb, lane);
        v16bf bm = ldBT(WxhT, KXH, nbase, kb, lane);
        acc = wmma_bf(a, bm, acc);
      }
      int n = nbase + (lane & 15);
      int mb = (lane >> 4) << 3;
#pragma unroll
      for (int r = 0; r < 8; ++r) {
        int m = mb + r;
        if (m < BEAM) s_gzr[m][n] = acc[r];
      }
    }
    // (h) xn (K=0..639) and hn (K=640..1151) for cols 1024..1535
    for (int nt = 0; nt < 4; ++nt) {
      int ct = wave * 4 + nt;
      int nbase = 1024 + ct * 16;
      v8f accX = {}, accH = {};
      for (int kb = 0; kb < 640; kb += 32) {
        v16bf a  = ldA(&s_xin[0][0], KXH, BEAM, kb, lane);
        v16bf bm = ldBT(WxhT, KXH, nbase, kb, lane);
        accX = wmma_bf(a, bm, accX);
      }
      for (int kb = 640; kb < KXH; kb += 32) {
        v16bf a  = ldA(&s_xin[0][0], KXH, BEAM, kb, lane);
        v16bf bm = ldBT(WxhT, KXH, nbase, kb, lane);
        accH = wmma_bf(a, bm, accH);
      }
      int n = ct * 16 + (lane & 15);
      int mb = (lane >> 4) << 3;
#pragma unroll
      for (int r = 0; r < 8; ++r) {
        int m = mb + r;
        if (m < BEAM) { s_xn[m][n] = accX[r]; s_hn[m][n] = accH[r]; }
      }
    }
    __syncthreads();

    // (i) GRU combine
    for (int i = tid; i < BEAM * NS; i += 256) {
      int j = i >> 9, c = i & 511;
      float z  = sigm(s_gzr[j][c] + bg[c]);
      float r  = sigm(s_gzr[j][512 + c] + bg[512 + c]);
      float nn = tanhf(s_xn[j][c] + bg[1024 + c] + r * s_hn[j][c]);
      s_nstate[j][c] = z * s_state[j][c] + (1.0f - z) * nn;
    }
    __syncthreads();
    for (int i = tid; i < BEAM * NS; i += 256) {
      int j = i >> 9, c = i & 511;
      s_nsbf[j][c] = (bf16_t)s_nstate[j][c];
    }
    __syncthreads();

    // (j) probs = sigmoid(ns @ Wc + bc), gated by first-step and finished
    for (int tt = wave; tt < NCLSP / 16; tt += 8) {
      int nbase = tt * 16;
      v8f acc = {};
      for (int kb = 0; kb < NS; kb += 32) {
        v16bf a  = ldA(&s_nsbf[0][0], NS, BEAM, kb, lane);
        v16bf bm = ldBT(WcT, NS, nbase, kb, lane);
        acc = wmma_bf(a, bm, acc);
      }
      int n = nbase + (lane & 15);
      int mb = (lane >> 4) << 3;
#pragma unroll
      for (int r = 0; r < 8; ++r) {
        int m = mb + r;
        if (m < BEAM) {
          float p = 0.0f;
          if (n < NCLS) {
            p = sigm(acc[r] + bc[n]);
            if (t == 0 && n > 0) p = 0.0f;
            if (s_fin[m]) p = 0.0f;
          }
          s_probs[m][n] = p;
        }
      }
    }
    __syncthreads();

    // (k) per-thread top-4 over BEAM*NCHARS candidates
    {
      float bv0 = -1e38f, bv1 = -1e38f, bv2 = -1e38f, bv3 = -1e38f;
      int bi0 = -1, bi1 = -1, bi2 = -1, bi3 = -1;
      for (int i = tid; i < BEAM * NCHARS; i += 256) {
        int j = i / NCHARS;
        int u = i - j * NCHARS;
        float han = s_probs[j][0];
        float v;
        if (u < NHAN) {
          int c   = u / 588;               // 21*28
          int rem = u - c * 588;
          int jg  = rem / 28;
          int o   = rem - jg * 28;
          float combo = (s_probs[j][1 + c] + s_probs[j][21 + jg] + s_probs[j][43 + o]) * (1.0f / 3.0f);
          v = 0.5f * (logclip(han) + logclip(combo));
        } else {
          int si = u - NHAN;
          v = 0.5f * (logclip(1.0f - han) + logclip(s_probs[j][72 + si]));
        }
        if (v > bv3) {
          if (v > bv0)      { bv3 = bv2; bi3 = bi2; bv2 = bv1; bi2 = bi1; bv1 = bv0; bi1 = bi0; bv0 = v; bi0 = i; }
          else if (v > bv1) { bv3 = bv2; bi3 = bi2; bv2 = bv1; bi2 = bi1; bv1 = v; bi1 = i; }
          else if (v > bv2) { bv3 = bv2; bi3 = bi2; bv2 = v; bi2 = i; }
          else              { bv3 = v; bi3 = i; }
        }
      }
      s_cval[tid * 4 + 0] = bv0; s_cidx[tid * 4 + 0] = bi0;
      s_cval[tid * 4 + 1] = bv1; s_cidx[tid * 4 + 1] = bi1;
      s_cval[tid * 4 + 2] = bv2; s_cidx[tid * 4 + 2] = bi2;
      s_cval[tid * 4 + 3] = bv3; s_cidx[tid * 4 + 3] = bi3;
    }
    __syncthreads();

    // merge + beam bookkeeping (one lane)
    if (tid == 0) {
      float tv[BEAM] = {-1e38f, -1e38f, -1e38f, -1e38f};
      int   ti[BEAM] = {-1, -1, -1, -1};
      for (int e = 0; e < 1024; ++e) {
        float v = s_cval[e];
        int  ix = s_cidx[e];
        if (ix < 0) continue;
        if (v > tv[3] || (v == tv[3] && (ti[3] < 0 || ix < ti[3]))) {
          int p = 3;
          while (p > 0 && (v > tv[p - 1] || (v == tv[p - 1] && ix < ti[p - 1]))) {
            tv[p] = tv[p - 1]; ti[p] = ti[p - 1]; --p;
          }
          tv[p] = v; ti[p] = ix;
        }
      }
      int ouni[BEAM], olen[BEAM], ofin[BEAM];
      float olp[BEAM];
      for (int j = 0; j < BEAM; ++j) { ouni[j] = s_uni[j]; olen[j] = s_len[j]; ofin[j] = s_fin[j]; olp[j] = s_logp[j]; }
      (void)ouni;
      for (int jn = 0; jn < BEAM; ++jn) {
        int idx  = ti[jn];
        int pb   = idx / NCHARS;
        int uraw = idx - pb * NCHARS;
        int nuni;
        if (uraw >= NHAN) {
          int si = uraw - NHAN;
          if (si > 99) si = 99;
          nuni = (si == 0) ? 10 : (32 + si);     // SPECIAL_CODES
        } else {
          nuni = uraw + 44032;                    // HANGUL_BASE
        }
        if (ofin[jn]) nuni = -1;
        ids_tr[((size_t)t * BATCH + b) * BEAM + jn] = pb;
        uni_tr[((size_t)t * BATCH + b) * BEAM + jn] = nuni;
        if (ofin[jn]) {
          s_pb[jn]  = -1;     // keep old state
          s_uni[jn] = -1;
          s_fin[jn] = 1;
        } else {
          s_pb[jn]   = pb;
          s_uni[jn]  = nuni;
          s_logp[jn] = olp[pb] + tv[jn];
          s_len[jn]  = olen[pb] + 1;
          s_fin[jn]  = (ofin[pb] || nuni == 10 || nuni == -1) ? 1 : 0;
        }
      }
    }
    __syncthreads();

    // state reorder
    for (int i = tid; i < BEAM * NS; i += 256) {
      int j = i >> 9, c = i & 511;
      int p = s_pb[j];
      if (p >= 0) s_state[j][c] = s_nstate[p][c];
    }
    __syncthreads();
  }

  if (tid < BEAM) {
    flogp[b * BEAM + tid] = s_logp[tid];
    flen[b * BEAM + tid]  = s_len[tid];
  }
}

// ------------------------------ finalize ------------------------------------

__global__ void k_final(const int* __restrict__ ids_tr, const int* __restrict__ uni_tr,
                        const float* __restrict__ flogp, const int* __restrict__ flen,
                        float* __restrict__ out) {
  int b = blockIdx.x;
  __shared__ int s_order[BEAM];
  if (threadIdx.x == 0) {
    float sc[BEAM];
    for (int j = 0; j < BEAM; ++j) {
      float len = (float)flen[b * BEAM + j];
      float pen = __powf(5.0f + len, 0.7f) / __powf(6.0f, 0.7f);
      sc[j] = flogp[b * BEAM + j] / pen;
    }
    bool used[BEAM] = {false, false, false, false};
    for (int slot = 0; slot < BEAM; ++slot) {
      int best = -1;
      for (int j = 0; j < BEAM; ++j)
        if (!used[j] && (best < 0 || sc[j] > sc[best])) best = j;
      used[best] = true;
      s_order[slot] = best;
      out[BATCH * BEAM * MAXLEN + b * BEAM + slot] = sc[best];
    }
  }
  __syncthreads();
  for (int i = threadIdx.x; i < BEAM * MAXLEN; i += blockDim.x) {
    int slot = i / MAXLEN, t = i - slot * MAXLEN;
    int j  = s_order[slot];
    int tn = (t + 1) % MAXLEN;
    int rolled = ids_tr[((size_t)tn * BATCH + b) * BEAM + j];
    int u      = uni_tr[((size_t)t * BATCH + b) * BEAM + rolled];
    out[((size_t)b * BEAM + slot) * MAXLEN + t] = (float)u;
  }
}

// ------------------------------ launcher ------------------------------------

extern "C" void kernel_launch(void* const* d_in, const int* in_sizes, int n_in,
                              void* d_out, int out_size, void* d_ws, size_t ws_size,
                              hipStream_t stream) {
  (void)in_sizes; (void)n_in; (void)out_size; (void)ws_size;
  const float* enc  = (const float*)d_in[0];
  const int*   mskg = (const int*)d_in[1];
  const float* embt = (const float*)d_in[2];
  const float* Wk   = (const float*)d_in[3];
  const float* bk   = (const float*)d_in[4];
  const float* Wq   = (const float*)d_in[5];
  const float* bq   = (const float*)d_in[6];
  const float* Wsv  = (const float*)d_in[7];
  const float* bsv  = (const float*)d_in[8];
  const float* Wx   = (const float*)d_in[9];
  const float* Wh   = (const float*)d_in[10];
  const float* bg   = (const float*)d_in[11];
  const float* Wc   = (const float*)d_in[12];
  const float* bc   = (const float*)d_in[13];

  char* ws = (char*)d_ws;
  size_t off = 0;
  auto take = [&](size_t bytes) {
    void* p = ws + off;
    off += (bytes + 255) & ~(size_t)255;
    return p;
  };
  bf16_t* enc_bf = (bf16_t*)take((size_t)BATCH * SLEN * ENC * 2);
  bf16_t* encT   = (bf16_t*)take((size_t)BATCH * SLEN * ENC * 2);
  bf16_t* WkT    = (bf16_t*)take((size_t)ATT * ENC * 2);
  bf16_t* WqT    = (bf16_t*)take((size_t)ATT * NS * 2);
  bf16_t* WxhT   = (bf16_t*)take((size_t)NG * KXH * 2);
  bf16_t* WcT    = (bf16_t*)take((size_t)NCLSP * NS * 2);
  float*  key    = (float*)take((size_t)BATCH * SLEN * ATT * 4);
  int*    ids_tr = (int*)take((size_t)MAXLEN * BATCH * BEAM * 4);
  int*    uni_tr = (int*)take((size_t)MAXLEN * BATCH * BEAM * 4);
  float*  flogp  = (float*)take((size_t)BATCH * BEAM * 4);
  int*    flen   = (int*)take((size_t)BATCH * BEAM * 4);

  k_cvt_enc<<<2048, 256, 0, stream>>>(enc, enc_bf, encT);
  k_transpose<<<128, 256, 0, stream>>>(Wk, WkT, ENC, ATT);
  k_transpose<<<128, 256, 0, stream>>>(Wq, WqT, NS, ATT);
  k_wxh<<<1024, 256, 0, stream>>>(Wx, Wh, WxhT);
  k_wc<<<128, 256, 0, stream>>>(Wc, WcT);
  k_keys<<<(BATCH * SLEN) / 16, 256, 0, stream>>>(enc_bf, WkT, bk, key);
  k_decode<<<BATCH, 256, 0, stream>>>(key, encT, embt, mskg, WqT, bq, Wsv, bsv,
                                      WxhT, bg, WcT, bc, ids_tr, uni_tr, flogp, flen);
  k_final<<<BATCH, 64, 0, stream>>>(ids_tr, uni_tr, flogp, flen, (float*)d_out);
}
